// RoPEAttention_15161234555341
// MI455X (gfx1250) — matrix operand order repaired and checked by reading, not compiled, not run on hardware
//
#include <hip/hip_runtime.h>
#include <math.h>
#include <stdint.h>

// ---------------------------------------------------------------------------
// Types for CDNA5 WMMA (wave32, 16x16x32 bf16 -> f32)
// ---------------------------------------------------------------------------
typedef __bf16 bf16_t;
typedef bf16_t v16bf __attribute__((ext_vector_type(16)));
typedef bf16_t v8bf  __attribute__((ext_vector_type(8)));
typedef float  v8f   __attribute__((ext_vector_type(8)));

__device__ __forceinline__ v8f wmma_bf16(v16bf a, v16bf b, v8f c) {
  // D = A(16x32) * B(32x16) + C, fp32 accumulate
  return __builtin_amdgcn_wmma_f32_16x16x32_bf16(
      /*neg_a=*/false, a, /*neg_b=*/false, b,
      /*c_mod=*/(short)0, c, /*reuse_a=*/false, /*reuse_b=*/false);
}

// A-operand fragment: lane (0..15 | 16..31) holds row M = lane&15.
// K striping per ISA: half 0 -> K {k0..k0+7, k0+16..k0+23}, half 1 -> +8.
__device__ __forceinline__ v16bf frag_a(const bf16_t* p, int row0, int ld,
                                        int k0, int lane) {
  int r = row0 + (lane & 15);
  int h = lane >> 4;
  const bf16_t* base = p + (size_t)r * ld + k0 + h * 8;
  v8bf lo = *(const v8bf*)(base);
  v8bf hi = *(const v8bf*)(base + 16);
  return __builtin_shufflevector(lo, hi, 0, 1, 2, 3, 4, 5, 6, 7,
                                 8, 9, 10, 11, 12, 13, 14, 15);
}

// B-operand fragment: lane&15 = column N; K striping matches A:
// half 0 -> K {k0..k0+7, k0+16..k0+23}, half 1 -> +8.
__device__ __forceinline__ v16bf frag_b(const bf16_t* p, int row0, int ld,
                                        int k0, int lane) {
  int r = row0 + (lane & 15);
  int h = lane >> 4;
  const bf16_t* base = p + (size_t)r * ld + k0 + h * 8;
  v8bf lo = *(const v8bf*)(base);
  v8bf hi = *(const v8bf*)(base + 16);
  return __builtin_shufflevector(lo, hi, 0, 1, 2, 3, 4, 5, 6, 7,
                                 8, 9, 10, 11, 12, 13, 14, 15);
}

// Async bf16 tile copy global->LDS (16B per lane), tracked on ASYNCcnt.
__device__ __forceinline__ void async_b128(bf16_t* lds_dst, const bf16_t* g) {
  uint32_t lds_addr = (uint32_t)(uintptr_t)lds_dst;  // LDS byte address
  asm volatile("global_load_async_to_lds_b128 %0, %1, off"
               :: "v"(lds_addr), "v"(g) : "memory");
}

// ---------------------------------------------------------------------------
// GEMM: C[M,N] = A[M,K](f32) * W[N,K]^T(f32) (+bias), via bf16 WMMA.
// Block tile 128x128, 8 waves (4x2) -> each wave 32x64 (2x4 WMMA tiles).
// ---------------------------------------------------------------------------
#define GTM 128
#define GTN 128
#define GTK 32
#define LDA 48  // 96B row stride: 16B- and 32B-aligned rows
#define LDB 48

__global__ __launch_bounds__(256) void gemm_bf16_wmma(
    const float* __restrict__ A, const float* __restrict__ W,
    float* __restrict__ C, const float* __restrict__ bias,
    int M, int N, int K) {
  __shared__ __align__(32) bf16_t sA[GTM][LDA];
  __shared__ __align__(32) bf16_t sB[GTN][LDB];

  const int bm = blockIdx.x * GTM;
  const int bn = blockIdx.y * GTN;
  const int tid = threadIdx.x;
  const int lane = tid & 31;
  const int wave = tid >> 5;
  const int wm = (wave >> 1) * 32;  // 0,32,64,96
  const int wn = (wave & 1) * 64;   // 0,64

  v8f acc[2][4];
#pragma unroll
  for (int i = 0; i < 2; ++i)
#pragma unroll
    for (int j = 0; j < 4; ++j) acc[i][j] = (v8f){};

  const int ra = tid >> 1;        // staging row (0..127), both A and B
  const int ca = (tid & 1) * 16;  // staging col base

  for (int k0 = 0; k0 < K; k0 += GTK) {
    // Stage A 128x32 and B 128x32 (f32 -> bf16)
    const float* srcA = A + (size_t)(bm + ra) * K + k0 + ca;
    const float* srcB = W + (size_t)(bn + ra) * K + k0 + ca;
#pragma unroll
    for (int i = 0; i < 4; ++i) {
      float4 fa = *(const float4*)(srcA + i * 4);
      float4 fb = *(const float4*)(srcB + i * 4);
      int c = ca + i * 4;
      sA[ra][c + 0] = (bf16_t)fa.x; sA[ra][c + 1] = (bf16_t)fa.y;
      sA[ra][c + 2] = (bf16_t)fa.z; sA[ra][c + 3] = (bf16_t)fa.w;
      sB[ra][c + 0] = (bf16_t)fb.x; sB[ra][c + 1] = (bf16_t)fb.y;
      sB[ra][c + 2] = (bf16_t)fb.z; sB[ra][c + 3] = (bf16_t)fb.w;
    }
    if (k0 + GTK < K) {  // prefetch next K tile (global_prefetch_b8)
      __builtin_prefetch(srcA + GTK, 0, 1);
      __builtin_prefetch(srcB + GTK, 0, 1);
    }
    __syncthreads();

    v16bf a0 = frag_a(&sA[0][0], wm, LDA, 0, lane);
    v16bf a1 = frag_a(&sA[0][0], wm + 16, LDA, 0, lane);
#pragma unroll
    for (int j = 0; j < 4; ++j) {
      v16bf b = frag_b(&sB[0][0], wn + j * 16, LDB, 0, lane);
      acc[0][j] = wmma_bf16(a0, b, acc[0][j]);
      acc[1][j] = wmma_bf16(a1, b, acc[1][j]);
    }
    __syncthreads();
  }

  // C/D layout: lane&15 = column, VGPR r = row r + 8*(lane>>4)
  const int col = lane & 15;
  const int rowoff = (lane >> 4) * 8;
#pragma unroll
  for (int i = 0; i < 2; ++i) {
#pragma unroll
    for (int j = 0; j < 4; ++j) {
      int m0 = bm + wm + i * 16;
      int n0 = bn + wn + j * 16;
      float bv = bias ? bias[n0 + col] : 0.0f;
      const v8f& a = acc[i][j];
#pragma unroll
      for (int r = 0; r < 8; ++r)
        C[(size_t)(m0 + rowoff + r) * N + n0 + col] = a[r] + bv;
    }
  }
}

// ---------------------------------------------------------------------------
// RoPE + scale + scatter: qkv[BN,3*1024] f32 -> q/k/v [B,H,N,64] bf16
// ---------------------------------------------------------------------------
__global__ __launch_bounds__(256) void rope_scatter(
    const float* __restrict__ qkv, bf16_t* __restrict__ qb,
    bf16_t* __restrict__ kb, bf16_t* __restrict__ vb) {
  int idx = blockIdx.x * blockDim.x + threadIdx.x;  // over B*N*1024
  if (idx >= 8192 * 1024) return;
  int d = idx & 63;
  int h = (idx >> 6) & 15;
  int row = idx >> 10;   // b*2048 + n
  int n = row & 2047;
  int b = row >> 11;

  const float* base = qkv + (size_t)row * 3072 + h * 64;
  float qv = base[d];
  float kv = base[1024 + d];
  float vv = base[2048 + d];

  int j = d & 31;  // inv_freq index
  float freq = __powf(10000.0f, -(float)j * (1.0f / 32.0f));
  float ang = (float)n * freq;
  float s, c;
  __sincosf(ang, &s, &c);

  float qo, ko;
  if (d < 32) {
    qo = qv * c - base[d + 32] * s;
    ko = kv * c - base[1024 + d + 32] * s;
  } else {
    qo = qv * c + base[d - 32] * s;
    ko = kv * c + base[1024 + d - 32] * s;
  }
  size_t o = (((size_t)(b * 16 + h) * 2048) + n) * 64 + d;
  qb[o] = (bf16_t)(qo * 0.125f);  // hd^-0.5
  kb[o] = (bf16_t)ko;
  vb[o] = (bf16_t)vv;
}

// ---------------------------------------------------------------------------
// Flash-style attention: grid = (B*H, N/128), 8 waves x 16 q-rows each.
// K chunk staged by ASYNC global->LDS copy; V staged transposed via VGPRs.
// ---------------------------------------------------------------------------
#define LDK 80  // sK row stride (160B, 32B multiple)
#define LDV 48  // sV / sP row stride (96B)

__global__ __launch_bounds__(256) void attn_wmma(
    const bf16_t* __restrict__ Q, const bf16_t* __restrict__ Kb,
    const bf16_t* __restrict__ Vb, float* __restrict__ O) {
  const int SEQ = 2048;
  const int bh = blockIdx.x;
  const int qbase = blockIdx.y * 128;
  const int tid = threadIdx.x, lane = tid & 31, wave = tid >> 5;

  const size_t head = (size_t)bh * SEQ * 64;
  const bf16_t* qh = Q + head;
  const bf16_t* kh = Kb + head;
  const bf16_t* vh = Vb + head;

  __shared__ __align__(32) bf16_t sK[32][LDK];      // [key][hd]
  __shared__ __align__(32) bf16_t sV[64][LDV];      // [hd][key] (transposed)
  __shared__ __align__(32) bf16_t sP[8][16][LDV];   // per-wave P tile

  const int qrow0 = qbase + wave * 16;
  // Q fragments live in registers for the whole key loop
  v16bf aq0 = frag_a(qh, qrow0, 64, 0, lane);
  v16bf aq1 = frag_a(qh, qrow0, 64, 32, lane);

  v8f o0 = {}, o1 = {}, o2 = {}, o3 = {};
  float mrow[8], lrow[8];
#pragma unroll
  for (int r = 0; r < 8; ++r) { mrow[r] = -1e30f; lrow[r] = 0.0f; }

  const int srow = tid >> 3;        // staging: key row 0..31
  const int scol = (tid & 7) * 8;   // staging: 8 hd elems

  for (int k0 = 0; k0 < SEQ; k0 += 32) {
    // K chunk: async DMA straight into LDS (bf16 raw copy, no VGPR bounce)
    async_b128(&sK[srow][scol], kh + (size_t)(k0 + srow) * 64 + scol);
    // V chunk: transpose through VGPRs
    v8bf vv = *(const v8bf*)(vh + (size_t)(k0 + srow) * 64 + scol);
#pragma unroll
    for (int i = 0; i < 8; ++i) sV[scol + i][srow] = vv[i];
    asm volatile("s_wait_asynccnt 0" ::: "memory");
    __syncthreads();

    // S = q * K^T : two 16-key tiles, K-dim = hd = 64 (2 WMMAs each)
    v8f s0 = {}, s1 = {};
    s0 = wmma_bf16(aq0, frag_b(&sK[0][0], 0, LDK, 0, lane), s0);
    s0 = wmma_bf16(aq1, frag_b(&sK[0][0], 0, LDK, 32, lane), s0);
    s1 = wmma_bf16(aq0, frag_b(&sK[0][0], 16, LDK, 0, lane), s1);
    s1 = wmma_bf16(aq1, frag_b(&sK[0][0], 16, LDK, 32, lane), s1);

    // Online softmax (rows live in VGPR index + lane-half; reduce over 16 lanes)
#pragma unroll
    for (int r = 0; r < 8; ++r) {
      float mx = fmaxf(s0[r], s1[r]);
      mx = fmaxf(mx, __shfl_xor(mx, 1, 16));
      mx = fmaxf(mx, __shfl_xor(mx, 2, 16));
      mx = fmaxf(mx, __shfl_xor(mx, 4, 16));
      mx = fmaxf(mx, __shfl_xor(mx, 8, 16));
      float mnew = fmaxf(mrow[r], mx);
      float corr = __expf(mrow[r] - mnew);
      float p0 = __expf(s0[r] - mnew);
      float p1 = __expf(s1[r] - mnew);
      float ps = p0 + p1;
      ps += __shfl_xor(ps, 1, 16);
      ps += __shfl_xor(ps, 2, 16);
      ps += __shfl_xor(ps, 4, 16);
      ps += __shfl_xor(ps, 8, 16);
      lrow[r] = lrow[r] * corr + ps;
      mrow[r] = mnew;
      o0[r] *= corr; o1[r] *= corr; o2[r] *= corr; o3[r] *= corr;
      int rm = r + 8 * (lane >> 4);
      int kc = lane & 15;
      sP[wave][rm][kc] = (bf16_t)p0;
      sP[wave][rm][16 + kc] = (bf16_t)p1;
    }
    asm volatile("s_wait_dscnt 0" ::: "memory");  // wave-local P RAW in LDS

    // O += P(16x32) * V(32x64): 4 hd tiles
    v16bf ap = frag_a(&sP[wave][0][0], 0, LDV, 0, lane);
    o0 = wmma_bf16(ap, frag_b(&sV[0][0], 0, LDV, 0, lane), o0);
    o1 = wmma_bf16(ap, frag_b(&sV[0][0], 16, LDV, 0, lane), o1);
    o2 = wmma_bf16(ap, frag_b(&sV[0][0], 32, LDV, 0, lane), o2);
    o3 = wmma_bf16(ap, frag_b(&sV[0][0], 48, LDV, 0, lane), o3);
    __syncthreads();
  }

  // Normalize and write attn-out as [B, N, H*64] (transpose folded in)
  const int col = lane & 15;
  const int half = lane >> 4;
  const int b = bh >> 4, h = bh & 15;
#pragma unroll
  for (int r = 0; r < 8; ++r) {
    float inv = 1.0f / lrow[r];
    int qm = qrow0 + r + 8 * half;
    float* dst = O + ((size_t)(b * 2048 + qm)) * 1024 + h * 64;
    dst[col] = o0[r] * inv;
    dst[16 + col] = o1[r] * inv;
    dst[32 + col] = o2[r] * inv;
    dst[48 + col] = o3[r] * inv;
  }
}

// ---------------------------------------------------------------------------
// Host launcher
// ---------------------------------------------------------------------------
extern "C" void kernel_launch(void* const* d_in, const int* in_sizes, int n_in,
                              void* d_out, int out_size, void* d_ws,
                              size_t ws_size, hipStream_t stream) {
  (void)in_sizes; (void)n_in; (void)out_size; (void)ws_size;
  const float* x      = (const float*)d_in[0];   // [4,2048,1024]
  const float* w_qkv  = (const float*)d_in[1];   // [3072,1024]
  const float* w_proj = (const float*)d_in[2];   // [1024,1024]
  const float* b_proj = (const float*)d_in[3];   // [1024]
  float* out = (float*)d_out;                    // [4,2048,1024]

  char* ws = (char*)d_ws;
  const size_t MB = 1024 * 1024;
  bf16_t* qb = (bf16_t*)(ws);                    // 16 MB
  bf16_t* kb = (bf16_t*)(ws + 16 * MB);          // 16 MB
  bf16_t* vb = (bf16_t*)(ws + 32 * MB);          // 16 MB
  float* qkv_raw = (float*)(ws + 48 * MB);       // 96 MB
  float* attn_out = qkv_raw;                     // reuse (qkv dead after rope)

  // 1) qkv = x @ w_qkv^T
  gemm_bf16_wmma<<<dim3(8192 / GTM, 3072 / GTN), 256, 0, stream>>>(
      x, w_qkv, qkv_raw, nullptr, 8192, 3072, 1024);
  // 2) RoPE + scale + scatter to [B,H,N,hd] bf16
  rope_scatter<<<(8192 * 1024) / 256, 256, 0, stream>>>(qkv_raw, qb, kb, vb);
  // 3) attention -> attn_out [B,N,D] f32
  attn_wmma<<<dim3(64, 2048 / 128), 256, 0, stream>>>(qb, kb, vb, attn_out);
  // 4) out = attn_out @ w_proj^T + b_proj
  gemm_bf16_wmma<<<dim3(8192 / GTM, 1024 / GTN), 256, 0, stream>>>(
      attn_out, w_proj, out, b_proj, 8192, 1024, 1024);
}
